// MultiHeadAttention_60292750901419
// MI455X (gfx1250) — compile-verified, matrix operand order checked by reading, not run on hardware
//
#include <hip/hip_runtime.h>
#include <math.h>
#include <stdint.h>

// Problem constants (B=1)
#define L   4096
#define D   1024
#define NH  16
#define HD  64
#define N3  3072
#define ATTN_SCALE 0.125f   // 1/sqrt(64)
#define LDS_STRIDE 72       // 64 + 8 pad elems -> conflict-free ds_load_b128
#define TILE_ELEMS (128 * LDS_STRIDE)

typedef __attribute__((ext_vector_type(8)))  float   v8f;
typedef __attribute__((ext_vector_type(8)))  __bf16  bf16x8;
typedef __attribute__((ext_vector_type(16))) __bf16  bf16x16;

// ---------------------------------------------------------------------------
// Fragment helpers.
// A-fragment (16x32 bf16, wave32): lane m = lane%16; half h = lane/16;
//   elem[i]   = A[m][h*8 + i]       (i = 0..7)
//   elem[8+i] = A[m][16 + h*8 + i]
// B-fragment (32x16) uses the identical pattern on B^T (lane = column N).
// ---------------------------------------------------------------------------
__device__ __forceinline__ bf16x16 load_frag(const __bf16* p, int h8) {
  bf16x8 lo = *(const bf16x8*)(p + h8);
  bf16x8 hi = *(const bf16x8*)(p + h8 + 16);
  bf16x16 r;
#pragma unroll
  for (int i = 0; i < 8; ++i) { r[i] = lo[i]; r[i + 8] = hi[i]; }
  return r;
}

__device__ __forceinline__ v8f zero8() {
  v8f z;
#pragma unroll
  for (int i = 0; i < 8; ++i) z[i] = 0.0f;
  return z;
}

__device__ __forceinline__ v8f wmma_bf16(bf16x16 a, bf16x16 b, v8f c) {
  // v_wmma_f32_16x16x32_bf16: (neg_a, A, neg_b, B, c_mod, C, reuse_a, reuse_b)
  return __builtin_amdgcn_wmma_f32_16x16x32_bf16(false, a, false, b, (short)0, c,
                                                 false, false);
}

// ---------------------------------------------------------------------------
// gfx1250 async global->LDS copy (ASYNCcnt-tracked, no VGPR round trip).
// VDST VGPR carries the per-lane LDS byte offset; VADDR the global address.
// Generic shared pointers map to LDS offsets via their low 32 bits.
// ---------------------------------------------------------------------------
__device__ __forceinline__ uint32_t lds_off(const void* p) {
  return (uint32_t)(uintptr_t)p;
}

__device__ __forceinline__ void async_copy_b128(uint32_t ldsByteOff,
                                                const void* gsrc) {
  asm volatile("global_load_async_to_lds_b128 %0, %1, off"
               :: "v"(ldsByteOff), "v"(gsrc)
               : "memory");
}

__device__ __forceinline__ void wait_async0() {
  asm volatile("s_wait_asynccnt 0" ::: "memory");
}

// ---------------------------------------------------------------------------
// fp32 -> bf16 converters (straight + transpose). Weights are pre-transposed
// once so WMMA operand staging is contiguous 16-byte transfers.
// ---------------------------------------------------------------------------
__global__ void cvt_bf16(const float* __restrict__ in, __bf16* __restrict__ out,
                         int n) {
  int i = blockIdx.x * blockDim.x + threadIdx.x;
  if (i < n) out[i] = (__bf16)in[i];
}

__global__ void cvt_bf16_T(const float* __restrict__ in, __bf16* __restrict__ out,
                           int R, int C) {   // out[c*R + r] = in[r*C + c]
  int i = blockIdx.x * blockDim.x + threadIdx.x;
  if (i < R * C) {
    int r = i / C, c = i % C;
    out[(size_t)c * R + r] = (__bf16)in[i];
  }
}

// ---------------------------------------------------------------------------
// Shared GEMM core: block computes 128(M) x 128(N); 8 waves, each a 32x64
// register tile (2x4 accumulators).  Double-buffered pipeline: async-DMA the
// next 64-deep A/B k-stage into the alternate LDS buffer while WMMAs consume
// the current one; one s_wait_asynccnt + one barrier per stage.
// lds layout: [buf(2)][A|B(2)][TILE_ELEMS]
// ---------------------------------------------------------------------------
__device__ __forceinline__ void gemm_core(
    const __bf16* __restrict__ A, const __bf16* __restrict__ Bt,
    int K, int blockM, int blockN, __bf16* lds, v8f acc[2][4]) {
  const int tid  = threadIdx.x;
  const int lane = tid & 31;
  const int w    = tid >> 5;
  const int ln = lane & 15;
  const int hf = lane >> 4;
  const int h8 = hf * 8;
  const int mloc = (w >> 1) * 32;   // wave M offset inside block (4 waves)
  const int nloc = (w & 1) * 64;    // wave N offset inside block (2 waves)
  const int srow = tid >> 1;        // staging: 2 threads per row
  const int scol = (tid & 1) * 32;  // each thread copies 32 elems (64 B)

#pragma unroll
  for (int mt = 0; mt < 2; ++mt)
#pragma unroll
    for (int nt = 0; nt < 4; ++nt) acc[mt][nt] = zero8();

  const __bf16* aSrc = A  + (size_t)(blockM + srow) * K + scol;
  const __bf16* bSrc = Bt + (size_t)(blockN + srow) * K + scol;
  const uint32_t stageOff = (uint32_t)(srow * LDS_STRIDE + scol) * 2u;
  uint32_t dstA[2], dstB[2];
  dstA[0] = lds_off(lds + 0 * TILE_ELEMS) + stageOff;
  dstB[0] = lds_off(lds + 1 * TILE_ELEMS) + stageOff;
  dstA[1] = lds_off(lds + 2 * TILE_ELEMS) + stageOff;
  dstB[1] = lds_off(lds + 3 * TILE_ELEMS) + stageOff;

  // Prologue: stage k0 = 0 into buffer 0.
#pragma unroll
  for (int i = 0; i < 4; ++i) {
    async_copy_b128(dstA[0] + i * 16, aSrc + i * 8);
    async_copy_b128(dstB[0] + i * 16, bSrc + i * 8);
  }
  wait_async0();
  __syncthreads();

  int cur = 0;
  for (int k0 = 0; k0 < K; k0 += 64) {
    // Prefetch next stage into the alternate buffer (overlaps with WMMAs).
    if (k0 + 64 < K) {
      const __bf16* aN = aSrc + (k0 + 64);
      const __bf16* bN = bSrc + (k0 + 64);
#pragma unroll
      for (int i = 0; i < 4; ++i) {
        async_copy_b128(dstA[cur ^ 1] + i * 16, aN + i * 8);
        async_copy_b128(dstB[cur ^ 1] + i * 16, bN + i * 8);
      }
    }

    const __bf16* ldsA = lds + (cur * 2 + 0) * TILE_ELEMS;
    const __bf16* ldsB = lds + (cur * 2 + 1) * TILE_ELEMS;
#pragma unroll
    for (int ks = 0; ks < 64; ks += 32) {
      bf16x16 a0 = load_frag(ldsA + (mloc + ln) * LDS_STRIDE + ks, h8);
      bf16x16 a1 = load_frag(ldsA + (mloc + 16 + ln) * LDS_STRIDE + ks, h8);
#pragma unroll
      for (int nt = 0; nt < 4; ++nt) {
        bf16x16 b = load_frag(ldsB + (nloc + nt * 16 + ln) * LDS_STRIDE + ks, h8);
        acc[0][nt] = wmma_bf16(a0, b, acc[0][nt]);
        acc[1][nt] = wmma_bf16(a1, b, acc[1][nt]);
      }
    }

    wait_async0();      // own prefetch landed in LDS
    __syncthreads();    // whole block's stage visible; reads of cur done
    cur ^= 1;
  }
}

// ---------------------------------------------------------------------------
// QKV GEMM: [L x D] @ [D x 3D] + bias.  Epilogue scatters to per-head Q, K
// and transposed V (bf16) so the attention kernel reads K-contiguous.
// ---------------------------------------------------------------------------
__global__ __launch_bounds__(256) void gemm_qkv(
    const __bf16* __restrict__ Xb, const __bf16* __restrict__ Wt,
    const float* __restrict__ bias, __bf16* __restrict__ Qb,
    __bf16* __restrict__ Kb, __bf16* __restrict__ Vt) {
  __shared__ __bf16 ldsBuf[4 * TILE_ELEMS];    // 72 KB, double-buffered A/B
  v8f acc[2][4];
  const int blockM = blockIdx.y * 128;
  const int blockN = blockIdx.x * 128;
  gemm_core(Xb, Wt, D, blockM, blockN, ldsBuf, acc);

  const int lane = threadIdx.x & 31;
  const int w    = threadIdx.x >> 5;
  const int ln = lane & 15;
  const int hf = lane >> 4;
  const int waveM = blockM + (w >> 1) * 32;
  const int waveN = blockN + (w & 1) * 64;

#pragma unroll
  for (int mt = 0; mt < 2; ++mt)
#pragma unroll
    for (int nt = 0; nt < 4; ++nt)
#pragma unroll
      for (int r = 0; r < 8; ++r) {
        int m = waveM + mt * 16 + r + 8 * hf;     // token index
        int c = waveN + nt * 16 + ln;             // column in [0, 3072)
        float val = acc[mt][nt][r] + bias[c];
        __bf16 bv = (__bf16)val;
        int p = c >> 10, rem = c & 1023;
        int hh = rem >> 6, dd = rem & 63;
        if (p == 0)      Qb[((size_t)hh * L + m) * HD + dd] = bv;
        else if (p == 1) Kb[((size_t)hh * L + m) * HD + dd] = bv;
        else             Vt[((size_t)hh * HD + dd) * L + m] = bv; // transposed
      }
}

// ---------------------------------------------------------------------------
// Output projection: [L x D](bf16) @ [D x D] + bias -> fp32 d_out.
// ---------------------------------------------------------------------------
__global__ __launch_bounds__(256) void gemm_out(
    const __bf16* __restrict__ A, const __bf16* __restrict__ Bt,
    const float* __restrict__ bias, float* __restrict__ Cout,
    int M, int N, int K) {
  __shared__ __bf16 ldsBuf[4 * TILE_ELEMS];
  v8f acc[2][4];
  const int blockM = blockIdx.y * 128;
  const int blockN = blockIdx.x * 128;
  gemm_core(A, Bt, K, blockM, blockN, ldsBuf, acc);

  const int lane = threadIdx.x & 31;
  const int w    = threadIdx.x >> 5;
  const int ln = lane & 15;
  const int hf = lane >> 4;
  const int waveM = blockM + (w >> 1) * 32;
  const int waveN = blockN + (w & 1) * 64;

#pragma unroll
  for (int mt = 0; mt < 2; ++mt)
#pragma unroll
    for (int nt = 0; nt < 4; ++nt)
#pragma unroll
      for (int r = 0; r < 8; ++r) {
        int m = waveM + mt * 16 + r + 8 * hf;
        int n = waveN + nt * 16 + ln;
        Cout[(size_t)m * N + n] = acc[mt][nt][r] + bias[n];
      }
}

// ---------------------------------------------------------------------------
// Flash attention, causal. One wave per 16-query tile; 4 independent waves
// per block (no barriers: each wave has its own causal trip count).
// Per 32-key step: 4 WMMA for S = Q K^T, online softmax (shfl_xor reductions
// inside 16-lane halves), P relayout via private LDS patch, 4 WMMA for P V.
// ---------------------------------------------------------------------------
__global__ __launch_bounds__(128) void attn_kernel(
    const __bf16* __restrict__ Qb, const __bf16* __restrict__ Kb,
    const __bf16* __restrict__ Vt, __bf16* __restrict__ Ob) {
  __shared__ __bf16 ldsP[4][16 * 32];          // per-wave P staging (bf16)
  const int lane  = threadIdx.x & 31;
  const int w     = threadIdx.x >> 5;
  const int hh    = blockIdx.y;                // head
  const int qbase = blockIdx.x * 64 + w * 16;  // query tile base
  const int ln = lane & 15;
  const int hf = lane >> 4;
  const int h8 = hf * 8;

  const __bf16* qh = Qb + (size_t)hh * L * HD;
  const __bf16* kh = Kb + (size_t)hh * L * HD;
  const __bf16* vh = Vt + (size_t)hh * HD * L;
  __bf16* myP = &ldsP[w][0];

  const __bf16* qrow = qh + (size_t)(qbase + ln) * HD;
  bf16x16 aq0 = load_frag(qrow, h8);           // K-dim 0..31
  bf16x16 aq1 = load_frag(qrow + 32, h8);      // K-dim 32..63

  v8f o[4];
#pragma unroll
  for (int j = 0; j < 4; ++j) o[j] = zero8();
  float mi[8], li[8];
#pragma unroll
  for (int r = 0; r < 8; ++r) { mi[r] = -__builtin_inff(); li[r] = 0.0f; }

  for (int kb = 0; kb < qbase + 16; kb += 32) {
    // ---- S = Q K^T for two 16-key subtiles (K^T B-frag == A-layout of K) --
    const __bf16* krow0 = kh + (size_t)(kb + ln) * HD;
    const __bf16* krow1 = kh + (size_t)(kb + 16 + ln) * HD;
    v8f s0 = zero8(), s1 = zero8();
    s0 = wmma_bf16(aq0, load_frag(krow0, h8), s0);
    s0 = wmma_bf16(aq1, load_frag(krow0 + 32, h8), s0);
    s1 = wmma_bf16(aq0, load_frag(krow1, h8), s1);
    s1 = wmma_bf16(aq1, load_frag(krow1 + 32, h8), s1);

    // ---- scale, causal mask, online softmax -----------------------------
#pragma unroll
    for (int r = 0; r < 8; ++r) {
      int qi = qbase + r + 8 * hf;
      float x0 = s0[r] * ATTN_SCALE;
      float x1 = s1[r] * ATTN_SCALE;
      if (kb + ln > qi)      x0 = -__builtin_inff();
      if (kb + 16 + ln > qi) x1 = -__builtin_inff();

      float mx = fmaxf(x0, x1);
      mx = fmaxf(mx, __shfl_xor(mx, 1, 32));
      mx = fmaxf(mx, __shfl_xor(mx, 2, 32));
      mx = fmaxf(mx, __shfl_xor(mx, 4, 32));
      mx = fmaxf(mx, __shfl_xor(mx, 8, 32));   // stays within 16-lane half
      float mnew  = fmaxf(mi[r], mx);
      float alpha = __expf(mi[r] - mnew);
      float p0 = __expf(x0 - mnew);
      float p1 = __expf(x1 - mnew);
      float rs = p0 + p1;
      rs += __shfl_xor(rs, 1, 32);
      rs += __shfl_xor(rs, 2, 32);
      rs += __shfl_xor(rs, 4, 32);
      rs += __shfl_xor(rs, 8, 32);
      li[r] = li[r] * alpha + rs;
      mi[r] = mnew;
#pragma unroll
      for (int j = 0; j < 4; ++j) o[j][r] *= alpha;

      int mloc = r + 8 * hf;                   // C-layout row of this elem
      myP[mloc * 32 + ln]      = (__bf16)p0;
      myP[mloc * 32 + 16 + ln] = (__bf16)p1;
    }
    asm volatile("s_wait_dscnt 0" ::: "memory");   // LDS stores -> loads

    // ---- P (16x32) as A-fragment; V B-frags contiguous via Vt -----------
    bf16x16 ap = load_frag(myP + ln * 32, h8);
#pragma unroll
    for (int j = 0; j < 4; ++j) {
      const __bf16* vrow = vh + (size_t)(j * 16 + ln) * L + kb;
      o[j] = wmma_bf16(ap, load_frag(vrow, h8), o[j]);
    }
  }

  // ---- normalize and emit bf16 [L x D] for the output projection --------
#pragma unroll
  for (int r = 0; r < 8; ++r) {
    float inv = 1.0f / li[r];
    int l = qbase + r + 8 * hf;
#pragma unroll
    for (int j = 0; j < 4; ++j)
      Ob[(size_t)l * D + hh * HD + j * 16 + ln] = (__bf16)(o[j][r] * inv);
  }
}

// ---------------------------------------------------------------------------
extern "C" void kernel_launch(void* const* d_in, const int* in_sizes, int n_in,
                              void* d_out, int out_size, void* d_ws,
                              size_t ws_size, hipStream_t stream) {
  const float* X     = (const float*)d_in[0];
  const float* Wqkv  = (const float*)d_in[1];
  const float* bqkv  = (const float*)d_in[2];
  const float* Wproj = (const float*)d_in[3];
  const float* bproj = (const float*)d_in[4];
  float* out = (float*)d_out;

  char* ws = (char*)d_ws;
  size_t off = 0;
  __bf16* Xb     = (__bf16*)(ws + off); off += (size_t)L * D * 2;    // 8 MB
  __bf16* WqkvT  = (__bf16*)(ws + off); off += (size_t)N3 * D * 2;   // 6 MB
  __bf16* WprojT = (__bf16*)(ws + off); off += (size_t)D * D * 2;    // 2 MB
  __bf16* Qb     = (__bf16*)(ws + off); off += (size_t)NH * L * HD * 2; // 8 MB
  __bf16* Kb     = (__bf16*)(ws + off); off += (size_t)NH * L * HD * 2; // 8 MB
  __bf16* Vt     = (__bf16*)(ws + off); off += (size_t)NH * HD * L * 2; // 8 MB
  __bf16* AttnB  = (__bf16*)(ws + off); off += (size_t)L * D * 2;    // 8 MB

  // 1) precision/layout staging (one-time, trivially bandwidth-cheap)
  int nX = L * D;
  cvt_bf16<<<(nX + 255) / 256, 256, 0, stream>>>(X, Xb, nX);
  int nW1 = D * N3;
  cvt_bf16_T<<<(nW1 + 255) / 256, 256, 0, stream>>>(Wqkv, WqkvT, D, N3);
  int nW2 = D * D;
  cvt_bf16_T<<<(nW2 + 255) / 256, 256, 0, stream>>>(Wproj, WprojT, D, D);

  // 2) QKV projection (async-LDS double-buffered WMMA) + Q/K/V^T scatter
  gemm_qkv<<<dim3(N3 / 128, L / 128), 256, 0, stream>>>(Xb, WqkvT, bqkv,
                                                        Qb, Kb, Vt);

  // 3) causal flash attention (WMMA + online softmax)
  attn_kernel<<<dim3(L / 64, NH), 128, 0, stream>>>(Qb, Kb, Vt, AttnB);

  // 4) output projection (async-LDS double-buffered WMMA), fp32 result
  gemm_out<<<dim3(D / 128, L / 128), 256, 0, stream>>>(AttnB, WprojT, bproj,
                                                       out, L, D, D);
}
